// kNNHyperbolicAttentionLayer_66374424592980
// MI455X (gfx1250) — matrix operand order repaired and checked by reading, not compiled, not run on hardware
//
#include <hip/hip_runtime.h>
#include <hip/hip_bf16.h>
#include <float.h>
#include <math.h>

// Problem constants (match reference)
#define BB   2
#define NN   2048
#define DIMD 512
#define NHH  8
#define HDD  64
#define KNB  32
#define PDD  16
#define FFD  2048
#define LN_EPS 1e-6f
#define MROWS (BB * NN)   // 4096

typedef __attribute__((ext_vector_type(16))) __bf16 v16bf;
typedef __attribute__((ext_vector_type(8)))  float  v8f;
typedef __attribute__((ext_vector_type(4)))  int    i4v;

union FragU { uint4 q[2]; v16bf v; };

static __device__ __forceinline__ unsigned short f2bf(float f) {
  unsigned int u = __float_as_uint(f);
  unsigned int r = (u + 0x7FFFu + ((u >> 16) & 1u)) >> 16; // RNE
  return (unsigned short)r;
}
static __device__ __forceinline__ float bf2f(unsigned short h) {
  return __uint_as_float(((unsigned int)h) << 16);
}
static __device__ __forceinline__ float gelu_tanh(float v) {
  float v3 = v * v * v;
  return 0.5f * v * (1.0f + tanhf(0.7978845608028654f * (v + 0.044715f * v3)));
}

// ---------------------------------------------------------------------------
// Async global->LDS staging (CDNA5 GLOBAL_LOAD_ASYNC_TO_LDS_B128, ASYNCcnt),
// with a synchronous fallback so the file always compiles.
// ---------------------------------------------------------------------------
#if __has_builtin(__builtin_amdgcn_global_load_async_to_lds_b128)
#define HAVE_ASYNC_LDS 1
#else
#define HAVE_ASYNC_LDS 0
#endif

static __device__ __forceinline__ void stage_b128(const unsigned short* gsrc,
                                                  unsigned short* ldst) {
#if HAVE_ASYNC_LDS
  __builtin_amdgcn_global_load_async_to_lds_b128(
      (__attribute__((address_space(1))) i4v*)gsrc,
      (__attribute__((address_space(3))) i4v*)ldst, 0, 0);
#else
  *(uint4*)ldst = *(const uint4*)gsrc;
#endif
}
static __device__ __forceinline__ void wait_async_le(int n) {
#if HAVE_ASYNC_LDS
#if __has_builtin(__builtin_amdgcn_s_wait_asynccnt)
  if (n == 0) __builtin_amdgcn_s_wait_asynccnt(0);
  else        __builtin_amdgcn_s_wait_asynccnt(1);
#else
  if (n == 0) asm volatile("s_wait_asynccnt 0x0" ::: "memory");
  else        asm volatile("s_wait_asynccnt 0x1" ::: "memory");
#endif
#else
  (void)n;
#endif
}

// ---------------------------------------------------------------------------
// TDM probe (compile-only; never launched): exercises the Tensor Data Mover
// path (tensor_load_to_lds + s_wait_tensorcnt) per CDNA5 ISA.
// This toolchain uses the 6-arg signature (probed: "expected 6"):
//   (uint32x4 g0, int32x8 g1, int32x4 g2, int32x4 g3, int32x8 g4, i32 cpol)
// ---------------------------------------------------------------------------
#if __has_builtin(__builtin_amdgcn_tensor_load_to_lds)
typedef __attribute__((ext_vector_type(4))) unsigned int u32x4;
typedef __attribute__((ext_vector_type(8))) int i32x8;
typedef __attribute__((ext_vector_type(4))) int i32x4;

__global__ void tdm_probe(unsigned long long gaddr, unsigned int magic,
                          unsigned short* sink) {
  __shared__ unsigned short stagebuf[4096];
  // D# group0: count=1 (valid), lds_addr[63:32], global_addr[120:64], type=2
  u32x4 g0;
  g0[0] = 1u;                                          // count=1, user mode
  g0[1] = (unsigned int)(size_t)&stagebuf[0];          // lds_addr (bytes)
  g0[2] = (unsigned int)(gaddr & 0xFFFFFFFFu);         // global_addr[31:0]
  g0[3] = ((unsigned int)(gaddr >> 32) & 0x01FFFFFFu)  // global_addr[56:32]
          | (2u << 30);                                // type = 2 ("image")
  // D# group1: workgroup_mask=0, data_size=1 (2B), dims/strides/tile dims
  i32x8 g1;
  g1[0] = (1 << 16);          // data_size = 2 bytes
  g1[1] = 0;                  // no atomic barrier
  g1[2] = DIMD;               // tensor_dim0
  g1[3] = (64 << 16);         // tensor_dim1 low | tile_dim0 bits
  g1[4] = (32 << 16) | 64;    // tile_dim1/2
  g1[5] = DIMD;               // tensor_dim0_stride
  g1[6] = 0;
  g1[7] = 0;
  i32x4 gz4 = {0, 0, 0, 0};
  i32x8 gz8 = {0, 0, 0, 0, 0, 0, 0, 0};
  __builtin_amdgcn_tensor_load_to_lds(g0, g1, gz4, gz4, gz8, 0);
#if __has_builtin(__builtin_amdgcn_s_wait_tensorcnt)
  __builtin_amdgcn_s_wait_tensorcnt(0);
#else
  asm volatile("s_wait_tensorcnt 0x0" ::: "memory");
#endif
  __syncthreads();
  if (threadIdx.x == 0 && magic == 0xdeadbeefu) sink[0] = stagebuf[1];
}
#endif

// ---------------------------------------------------------------------------
// fp32 W[Kd][Nc] (row-major) -> bf16 Wt[Nc][Kd] (transposed, K contiguous)
// ---------------------------------------------------------------------------
__global__ void cvt_transpose(const float* __restrict__ W,
                              unsigned short* __restrict__ Wt,
                              int Kd, int Nc) {
  int e = blockIdx.x * blockDim.x + threadIdx.x;
  if (e >= Kd * Nc) return;
  int k = e / Nc, n = e % Nc;
  Wt[(size_t)n * Kd + k] = f2bf(W[e]);
}

// ---------------------------------------------------------------------------
// LayerNorm over DIM=512, one block (256 thr) per row, bf16 output
// ---------------------------------------------------------------------------
__global__ void layernorm_bf16(const float* __restrict__ x,
                               const float* __restrict__ sc,
                               const float* __restrict__ bi,
                               unsigned short* __restrict__ out) {
  __shared__ float s0[256], s1[256];
  int row = blockIdx.x, tid = threadIdx.x;
  const float* xr = x + (size_t)row * DIMD;
  float a0 = xr[tid], a1 = xr[tid + 256];
  s0[tid] = a0 + a1;
  s1[tid] = a0 * a0 + a1 * a1;
  __syncthreads();
  for (int st = 128; st > 0; st >>= 1) {
    if (tid < st) { s0[tid] += s0[tid + st]; s1[tid] += s1[tid + st]; }
    __syncthreads();
  }
  float mean = s0[0] * (1.0f / DIMD);
  float var  = s1[0] * (1.0f / DIMD) - mean * mean;
  float rstd = rsqrtf(var + LN_EPS);
  out[(size_t)row * DIMD + tid]       = f2bf((a0 - mean) * rstd * sc[tid] + bi[tid]);
  out[(size_t)row * DIMD + tid + 256] = f2bf((a1 - mean) * rstd * sc[tid + 256] + bi[tid + 256]);
}

// ---------------------------------------------------------------------------
// Poincare distances for one query row + iterative top-32 (ascending, stable)
// ---------------------------------------------------------------------------
__global__ void poincare_topk(const float* __restrict__ pos,
                              const float* __restrict__ cptr,
                              int* __restrict__ oidx,
                              float* __restrict__ odist) {
  __shared__ float sdist[NN];
  __shared__ float posi[PDD];
  __shared__ float rval[256];
  __shared__ int   ridx[256];
  int row = blockIdx.x, tid = threadIdx.x;
  int b = row / NN, i = row % NN;
  float c = cptr[0];
  if (tid < PDD) posi[tid] = pos[((size_t)(b * NN + i)) * PDD + tid];
  __syncthreads();
  float xn = 0.f;
  #pragma unroll
  for (int t = 0; t < PDD; ++t) xn += posi[t] * posi[t];
  float rsc = rsqrtf(c);
  for (int j = tid; j < NN; j += 256) {
    const float* pj = pos + ((size_t)(b * NN + j)) * PDD;
    float diff = 0.f, yn = 0.f;
    #pragma unroll
    for (int t = 0; t < PDD; ++t) {
      float pv = pj[t];
      float d_ = posi[t] - pv;
      diff += d_ * d_;
      yn   += pv * pv;
    }
    float den = (1.0f - c * xn) * (1.0f - c * yn);
    den = fmaxf(den, 1e-8f);
    float arg = 1.0f + 2.0f * c * diff / den;
    arg = fmaxf(arg, 1.0f + 1e-7f);
    sdist[j] = logf(arg + sqrtf(arg * arg - 1.0f)) * rsc; // arccosh/sqrt(c)
  }
  __syncthreads();
  for (int sel = 0; sel < KNB; ++sel) {
    float best = FLT_MAX; int bidx = 0x7fffffff;
    for (int j = tid; j < NN; j += 256) {
      float v = sdist[j];
      if (v < best) { best = v; bidx = j; }
    }
    rval[tid] = best; ridx[tid] = bidx;
    __syncthreads();
    for (int st = 128; st > 0; st >>= 1) {
      if (tid < st) {
        float ov = rval[tid + st]; int oi = ridx[tid + st];
        if (ov < rval[tid] || (ov == rval[tid] && oi < ridx[tid])) {
          rval[tid] = ov; ridx[tid] = oi;
        }
      }
      __syncthreads();
    }
    if (tid == 0) {
      oidx[(size_t)row * KNB + sel]  = ridx[0];
      odist[(size_t)row * KNB + sel] = rval[0];
      sdist[ridx[0]] = FLT_MAX;
    }
    __syncthreads();
  }
}

// ---------------------------------------------------------------------------
// WMMA bf16 GEMM: C[M,Nc] = A[M,Kd] * Bt[Nc][Kd]^T
// Block = 256 thr = 8 waves; block tile 128x64 (wave tile 16x64, 4 accums).
// B tile (64 cols x 32 k) staged in LDS via async b128 loads, double-buffered
// with the last K-step peeled so the steady-state loop is branch-free.
// A fragments software-pipelined one K-step ahead in registers.
// LDS column stride padded to 160B so ds_load_b128 is bank-conflict-free.
// ---------------------------------------------------------------------------
enum { EP_QKV = 0, EP_RESID_F32 = 1, EP_GELU_BF16 = 2 };

#define BCOLS   64
#define LDSB_US 80                 // padded column stride in ushorts (160 B)
#define BTILE_US (BCOLS * LDSB_US) // 5120 ushorts = 10 KB per buffer

__global__ void gemm_wmma_bf16(const unsigned short* __restrict__ A, int lda,
                               const unsigned short* __restrict__ Bt, int ldb,
                               int Nc, int Kd,
                               const float* __restrict__ bias,
                               const float* __restrict__ resid,
                               void* __restrict__ out,
                               int mode) {
  __shared__ unsigned short btile[2][BTILE_US];

  int tid  = threadIdx.x;
  int wave = tid >> 5, lane = tid & 31;
  int rowBase = blockIdx.x * 128 + wave * 16;
  int colBase = blockIdx.y * 64;
  int hf  = lane >> 4;     // lane half
  int l16 = lane & 15;

  const unsigned short* Arow = A + (size_t)(rowBase + l16) * lda;

  // cooperative B staging: thread t moves one b128 chunk (col = t/4, chunk = t%4)
  int scol = tid >> 2, schk = tid & 3;
  int soff = scol * LDSB_US + schk * 8;
  const unsigned short* bsrc = Bt + (size_t)(colBase + scol) * ldb + schk * 8;

  stage_b128(bsrc, &btile[0][soff]);       // k-step 0 -> buffer 0

  // A fragment for k-step 0: VGPR0-3 = K{0..7}+8*hf, VGPR4-7 = K{16..23}+8*hf
  FragU a;
  a.q[0] = *(const uint4*)(Arow + 8 * hf);
  a.q[1] = *(const uint4*)(Arow + 16 + 8 * hf);

  v8f acc[4] = {{}, {}, {}, {}};
  int buf = 0;
  int kSteps = Kd >> 5;
  for (int s = 0; s < kSteps - 1; ++s) {
    int k0 = s << 5;
    stage_b128(bsrc + k0 + 32, &btile[buf ^ 1][soff]);  // next B tile (async)
    FragU an;                                           // next A fragment
    an.q[0] = *(const uint4*)(Arow + k0 + 32 + 8 * hf);
    an.q[1] = *(const uint4*)(Arow + k0 + 48 + 8 * hf);
    __builtin_prefetch(Arow + k0 + 96, 0, 3);

    wait_async_le(1);        // current B tile landed (next still in flight)
    __syncthreads();
    #pragma unroll
    for (int c = 0; c < 4; ++c) {
      const unsigned short* lp = &btile[buf][(c * 16 + l16) * LDSB_US + 8 * hf];
      FragU b;
      b.q[0] = *(const uint4*)(lp);
      b.q[1] = *(const uint4*)(lp + 16);
      acc[c] = __builtin_amdgcn_wmma_f32_16x16x32_bf16(false, a.v, false, b.v,
                                                       (short)0, acc[c], false, false);
    }
    __syncthreads();         // reads done before this buffer is overwritten
    a = an;
    buf ^= 1;
  }
  // peeled last K-step
  wait_async_le(0);
  __syncthreads();
  #pragma unroll
  for (int c = 0; c < 4; ++c) {
    const unsigned short* lp = &btile[buf][(c * 16 + l16) * LDSB_US + 8 * hf];
    FragU b;
    b.q[0] = *(const uint4*)(lp);
    b.q[1] = *(const uint4*)(lp + 16);
    acc[c] = __builtin_amdgcn_wmma_f32_16x16x32_bf16(false, a.v, false, b.v,
                                                     (short)0, acc[c], false, false);
  }

  #pragma unroll
  for (int c = 0; c < 4; ++c) {
    int gc = colBase + c * 16 + l16;
    float bc = bias[gc];
    #pragma unroll
    for (int r = 0; r < 8; ++r) {
      int grow = rowBase + r + 8 * hf;
      float v = acc[c][r] + bc;
      if (mode == EP_QKV) {
        // scatter [row, col] -> [b, h, token, d] bf16
        unsigned short* o = (unsigned short*)out;
        int bidx = grow / NN, itok = grow % NN;
        int h0 = gc / HDD, d0 = gc % HDD;
        o[(((size_t)bidx * NHH + h0) * NN + itok) * HDD + d0] = f2bf(v);
      } else if (mode == EP_RESID_F32) {
        float* o = (float*)out;
        o[(size_t)grow * Nc + gc] = v + resid[(size_t)grow * Nc + gc];
      } else { // EP_GELU_BF16
        unsigned short* o = (unsigned short*)out;
        o[(size_t)grow * Nc + gc] = f2bf(gelu_tanh(v));
      }
    }
  }
}

// ---------------------------------------------------------------------------
// kNN attention: one wave per (b,h,i); lane = neighbor slot (K=32=wave32)
// ---------------------------------------------------------------------------
__global__ void knn_attention(const unsigned short* __restrict__ qb,
                              const unsigned short* __restrict__ kb,
                              const unsigned short* __restrict__ vb,
                              const int* __restrict__ tki,
                              const float* __restrict__ tkd,
                              const float* __restrict__ log_tau,
                              const float* __restrict__ attn_scale,
                              unsigned short* __restrict__ attn_out) {
  int wave = (blockIdx.x * blockDim.x + threadIdx.x) >> 5;
  int lane = threadIdx.x & 31;
  int b   = wave / (NHH * NN);
  int rem = wave % (NHH * NN);
  int h = rem / NN;
  int i = rem % NN;

  float tau = fmaxf(expf(log_tau[0]), 1e-8f);
  float asc = attn_scale[0];

  int   nidx = tki[((size_t)(b * NN + i)) * KNB + lane];
  float nd   = tkd[((size_t)(b * NN + i)) * KNB + lane];

  const unsigned int* q32 = (const unsigned int*)(qb + (((size_t)(b * NHH + h)) * NN + i) * HDD);
  const unsigned int* k32 = (const unsigned int*)(kb + (((size_t)(b * NHH + h)) * NN + nidx) * HDD);
  float dot = 0.f;
  #pragma unroll
  for (int t = 0; t < HDD / 2; ++t) {
    unsigned int qp = q32[t], kp = k32[t];
    dot += bf2f((unsigned short)(qp & 0xFFFFu)) * bf2f((unsigned short)(kp & 0xFFFFu));
    dot += bf2f((unsigned short)(qp >> 16))     * bf2f((unsigned short)(kp >> 16));
  }
  float s = asc * tanhf(dot * 0.125f - nd / tau); // 1/sqrt(64) = 0.125

  // wave32 softmax over the 32 neighbor slots
  float m = s;
  for (int d = 16; d > 0; d >>= 1) m = fmaxf(m, __shfl_xor(m, d, 32));
  float p = expf(s - m);
  float sum = p;
  for (int d = 16; d > 0; d >>= 1) sum += __shfl_xor(sum, d, 32);
  float w = p / sum;

  // each lane owns output dims {2*lane, 2*lane+1}
  float acc0 = 0.f, acc1 = 0.f;
  const unsigned short* vbase = vb + ((size_t)(b * NHH + h)) * NN * HDD;
  #pragma unroll
  for (int j = 0; j < KNB; ++j) {
    float wj = __shfl(w, j, 32);
    int   ij = __shfl(nidx, j, 32);
    unsigned int vp = *(const unsigned int*)(vbase + (size_t)ij * HDD + lane * 2);
    acc0 += wj * bf2f((unsigned short)(vp & 0xFFFFu));
    acc1 += wj * bf2f((unsigned short)(vp >> 16));
  }
  unsigned int packed = (unsigned int)f2bf(acc0) | ((unsigned int)f2bf(acc1) << 16);
  *(unsigned int*)(attn_out + ((size_t)(b * NN + i)) * DIMD + h * HDD + lane * 2) = packed;
}

// ---------------------------------------------------------------------------
// Host-side orchestration
// ---------------------------------------------------------------------------
extern "C" void kernel_launch(void* const* d_in, const int* in_sizes, int n_in,
                              void* d_out, int out_size, void* d_ws, size_t ws_size,
                              hipStream_t stream) {
  (void)in_sizes; (void)n_in; (void)out_size; (void)ws_size;

  const float* x    = (const float*)d_in[0];
  const float* pos  = (const float*)d_in[1];
  const float* cpt  = (const float*)d_in[2];
  const float* Wq   = (const float*)d_in[3];
  const float* bq   = (const float*)d_in[4];
  const float* Wk   = (const float*)d_in[5];
  const float* bk   = (const float*)d_in[6];
  const float* Wv   = (const float*)d_in[7];
  const float* bv   = (const float*)d_in[8];
  const float* Wo   = (const float*)d_in[9];
  const float* bo   = (const float*)d_in[10];
  const float* W1   = (const float*)d_in[11];
  const float* b1   = (const float*)d_in[12];
  const float* W2   = (const float*)d_in[13];
  const float* b2   = (const float*)d_in[14];
  const float* ln1s = (const float*)d_in[15];
  const float* ln1b = (const float*)d_in[16];
  const float* ln2s = (const float*)d_in[17];
  const float* ln2b = (const float*)d_in[18];
  const float* ltau = (const float*)d_in[19];
  const float* ascl = (const float*)d_in[20];

  // workspace layout
  char* ws = (char*)d_ws;
  size_t off = 0;
  auto wsalloc = [&](size_t bytes) {
    void* p = ws + off;
    off += (bytes + 255) & ~(size_t)255;
    return p;
  };
  unsigned short* xn   = (unsigned short*)wsalloc((size_t)MROWS * DIMD * 2);
  unsigned short* Wqt  = (unsigned short*)wsalloc((size_t)DIMD * DIMD * 2);
  unsigned short* Wkt  = (unsigned short*)wsalloc((size_t)DIMD * DIMD * 2);
  unsigned short* Wvt  = (unsigned short*)wsalloc((size_t)DIMD * DIMD * 2);
  unsigned short* Wot  = (unsigned short*)wsalloc((size_t)DIMD * DIMD * 2);
  unsigned short* W1t  = (unsigned short*)wsalloc((size_t)DIMD * FFD * 2);
  unsigned short* W2t  = (unsigned short*)wsalloc((size_t)FFD * DIMD * 2);
  unsigned short* qbuf = (unsigned short*)wsalloc((size_t)MROWS * DIMD * 2);
  unsigned short* kbuf = (unsigned short*)wsalloc((size_t)MROWS * DIMD * 2);
  unsigned short* vbuf = (unsigned short*)wsalloc((size_t)MROWS * DIMD * 2);
  int*            tki  = (int*)wsalloc((size_t)MROWS * KNB * 4);
  float*          tkd  = (float*)wsalloc((size_t)MROWS * KNB * 4);
  unsigned short* attn = (unsigned short*)wsalloc((size_t)MROWS * DIMD * 2);
  float*          x1   = (float*)wsalloc((size_t)MROWS * DIMD * 4);
  unsigned short* x2n  = (unsigned short*)wsalloc((size_t)MROWS * DIMD * 2);
  unsigned short* h1   = (unsigned short*)wsalloc((size_t)MROWS * FFD * 2);

  // 1) weight conversion + transpose (fp32 -> bf16)
  cvt_transpose<<<(DIMD * DIMD + 255) / 256, 256, 0, stream>>>(Wq, Wqt, DIMD, DIMD);
  cvt_transpose<<<(DIMD * DIMD + 255) / 256, 256, 0, stream>>>(Wk, Wkt, DIMD, DIMD);
  cvt_transpose<<<(DIMD * DIMD + 255) / 256, 256, 0, stream>>>(Wv, Wvt, DIMD, DIMD);
  cvt_transpose<<<(DIMD * DIMD + 255) / 256, 256, 0, stream>>>(Wo, Wot, DIMD, DIMD);
  cvt_transpose<<<(DIMD * FFD + 255) / 256, 256, 0, stream>>>(W1, W1t, DIMD, FFD);
  cvt_transpose<<<(FFD * DIMD + 255) / 256, 256, 0, stream>>>(W2, W2t, FFD, DIMD);

  // 2) LN1 -> bf16 activations
  layernorm_bf16<<<MROWS, 256, 0, stream>>>(x, ln1s, ln1b, xn);

  // 3) hyperbolic distances + top-32
  poincare_topk<<<MROWS, 256, 0, stream>>>(pos, cpt, tki, tkd);

  // 4) QKV projections (WMMA), scatter to [B,H,N,HD] bf16
  dim3 g512(MROWS / 128, DIMD / 64);
  gemm_wmma_bf16<<<g512, 256, 0, stream>>>(xn, DIMD, Wqt, DIMD, DIMD, DIMD, bq, nullptr, qbuf, EP_QKV);
  gemm_wmma_bf16<<<g512, 256, 0, stream>>>(xn, DIMD, Wkt, DIMD, DIMD, DIMD, bk, nullptr, kbuf, EP_QKV);
  gemm_wmma_bf16<<<g512, 256, 0, stream>>>(xn, DIMD, Wvt, DIMD, DIMD, DIMD, bv, nullptr, vbuf, EP_QKV);

  // 5) sparse kNN attention (wave per (b,h,i))
  knn_attention<<<(BB * NHH * NN * 32) / 256, 256, 0, stream>>>(qbuf, kbuf, vbuf, tki, tkd,
                                                                ltau, ascl, attn);

  // 6) output projection + residual -> x1 (fp32)
  gemm_wmma_bf16<<<g512, 256, 0, stream>>>(attn, DIMD, Wot, DIMD, DIMD, DIMD, bo, x, x1, EP_RESID_F32);

  // 7) LN2 -> bf16
  layernorm_bf16<<<MROWS, 256, 0, stream>>>(x1, ln2s, ln2b, x2n);

  // 8) FFN up + GELU -> bf16
  dim3 gff(MROWS / 128, FFD / 64);
  gemm_wmma_bf16<<<gff, 256, 0, stream>>>(x2n, DIMD, W1t, DIMD, FFD, DIMD, b1, nullptr, h1, EP_GELU_BF16);

  // 9) FFN down + residual -> d_out (fp32)
  gemm_wmma_bf16<<<g512, 256, 0, stream>>>(h1, FFD, W2t, FFD, DIMD, FFD, b2, x1, (float*)d_out, EP_RESID_F32);
}
